// EGCN_7696581394471
// MI455X (gfx1250) — matrix operand order repaired and checked by reading, not compiled, not run on hardware
//
#include <hip/hip_runtime.h>
#include <math.h>

typedef __attribute__((ext_vector_type(16))) __bf16 v16bf;
typedef __attribute__((ext_vector_type(8)))  __bf16 v8bf;
typedef __attribute__((ext_vector_type(8)))  float  v8f;

__device__ __forceinline__ __bf16 f2bf(float f) {
    // round-to-nearest-even f32 -> bf16
    unsigned u = __builtin_bit_cast(unsigned, f);
    unsigned r = u + 0x7fffu + ((u >> 16) & 1u);
    unsigned short h = (unsigned short)(r >> 16);
    return __builtin_bit_cast(__bf16, h);
}

// ---------------- scorer norm ----------------
__global__ void invnorm_k(const float* __restrict__ s, float* __restrict__ o) {
    __shared__ float red[128];
    int t = threadIdx.x;
    float v = s[t];
    red[t] = v * v;
    __syncthreads();
    for (int off = 64; off > 0; off >>= 1) {
        if (t < off) red[t] += red[t + off];
        __syncthreads();
    }
    if (t == 0) o[0] = rsqrtf(red[0]);
}

// ---------------- node scores: X @ scorer * invnorm + mask ----------------
__global__ void score_k(const float* __restrict__ X, const float* __restrict__ sc,
                        const float* __restrict__ invn, const float* __restrict__ mask,
                        float* __restrict__ scores, int Nn) {
    int n = blockIdx.x * blockDim.x + threadIdx.x;
    if (n >= Nn) return;
    const float* xr = X + (size_t)n * 128;
    float acc = 0.f;
#pragma unroll 8
    for (int j = 0; j < 128; ++j) acc = fmaf(xr[j], sc[j], acc);
    scores[n] = acc * invn[0] + mask[n];
}

// ---------------- top-k via segment tournament (single block, 1024 thr) ----------------
__global__ void topk_k(float* __restrict__ scores, float* __restrict__ vals,
                       int* __restrict__ oidx, int Nn, int K) {
    __shared__ float sm[1024];
    __shared__ int   si[1024];
    __shared__ float rv[1024];
    __shared__ int   ri[1024];
    int tid = threadIdx.x;
    int chunk = (Nn + 1023) >> 10;
    int beg = tid * chunk;
    int end = beg + chunk; if (end > Nn) end = Nn;
    float mv = -INFINITY; int mi = -1;
    for (int j = beg; j < end; ++j) { float s = scores[j]; if (s > mv) { mv = s; mi = j; } }
    sm[tid] = mv; si[tid] = mi;
    __syncthreads();
    for (int i = 0; i < K; ++i) {
        rv[tid] = sm[tid]; ri[tid] = si[tid];
        __syncthreads();
        for (int off = 512; off > 0; off >>= 1) {
            if (tid < off) {
                float a = rv[tid], b = rv[tid + off];
                int ai = ri[tid], bi = ri[tid + off];
                bool take = false;
                if (bi >= 0) {
                    if (ai < 0) take = true;
                    else if (b > a) take = true;
                    else if (b == a && bi < ai) take = true; // stable: lower index first
                }
                if (take) { rv[tid] = b; ri[tid] = bi; }
            }
            __syncthreads();
        }
        int gi = ri[0]; float gv = rv[0];
        if (tid == 0) { oidx[i] = gi; vals[i] = gv; }
        int owner = gi / chunk;
        if (tid == owner) {
            scores[gi] = -INFINITY;
            float nv = -INFINITY; int ni = -1;
            for (int j = beg; j < end; ++j) { float s = scores[j]; if (s > nv) { nv = s; ni = j; } }
            sm[tid] = nv; si[tid] = ni;
        }
        __syncthreads();
    }
}

// ---------------- z[r,j] = X[idx[j], r] * tanh(vals[j]) ----------------
__global__ void zbuild_k(const float* __restrict__ X, const int* __restrict__ idx,
                         const float* __restrict__ vals, float* __restrict__ z, int K) {
    int i = blockIdx.x * blockDim.x + threadIdx.x;
    int total = 128 * K;
    if (i >= total) return;
    int r = i / K;
    int j = i - r * K;
    z[i] = X[(size_t)idx[j] * 128 + r] * tanhf(vals[j]);
}

// ---------------- matrix GRU, one block per column; also emits bf16 Q^T ----------------
__global__ void gru_k(const float* __restrict__ z, const float* __restrict__ Q,
                      const float* __restrict__ Wu, const float* __restrict__ Uu, const float* __restrict__ Bu,
                      const float* __restrict__ Wr, const float* __restrict__ Ur, const float* __restrict__ Br,
                      const float* __restrict__ Wh, const float* __restrict__ Uh, const float* __restrict__ Bh,
                      float* __restrict__ Qn, __bf16* __restrict__ QbfT, int cols) {
    __shared__ float zc[128], Qc[128], rq[128];
    int r = threadIdx.x;
    int c = blockIdx.x;
    zc[r] = z[(size_t)r * cols + c];
    Qc[r] = Q[(size_t)r * cols + c];
    __syncthreads();
    float au = Bu[(size_t)r * cols + c];
    float ar = Br[(size_t)r * cols + c];
    const float* wu = Wu + (size_t)r * 128; const float* uu = Uu + (size_t)r * 128;
    const float* wr = Wr + (size_t)r * 128; const float* ur = Ur + (size_t)r * 128;
#pragma unroll 4
    for (int j = 0; j < 128; ++j) {
        float zj = zc[j], qj = Qc[j];
        au = fmaf(wu[j], zj, au); au = fmaf(uu[j], qj, au);
        ar = fmaf(wr[j], zj, ar); ar = fmaf(ur[j], qj, ar);
    }
    float u  = 1.f / (1.f + __expf(-au));
    float rr = 1.f / (1.f + __expf(-ar));
    rq[r] = rr * Qc[r];
    __syncthreads();
    float ah = Bh[(size_t)r * cols + c];
    const float* wh = Wh + (size_t)r * 128; const float* uh = Uh + (size_t)r * 128;
#pragma unroll 4
    for (int j = 0; j < 128; ++j) {
        ah = fmaf(wh[j], zc[j], ah);
        ah = fmaf(uh[j], rq[j], ah);
    }
    float h = tanhf(ah);
    float qn = (1.f - u) * Qc[r] + u * h;
    Qn[(size_t)r * cols + c] = qn;
    QbfT[(size_t)c * 128 + r] = f2bf(qn);   // transposed bf16 copy for WMMA B-fragments
}

// ---------------- streaming f32 -> bf16 convert (one pass per layer input) ----------------
__global__ void cvtbf_k(const float* __restrict__ in, __bf16* __restrict__ outp, long n8) {
    long i = (long)blockIdx.x * blockDim.x + threadIdx.x;
    if (i >= n8) return;
    const float4* p = (const float4*)(in + i * 8);
    float4 a = p[0], b = p[1];
    v8bf o;
    o[0] = f2bf(a.x); o[1] = f2bf(a.y); o[2] = f2bf(a.z); o[3] = f2bf(a.w);
    o[4] = f2bf(b.x); o[5] = f2bf(b.y); o[6] = f2bf(b.z); o[7] = f2bf(b.w);
    *(v8bf*)(outp + i * 8) = o;
}

// ---------------- XW = X @ Qn via v_wmma_f32_16x16x32_bf16 ----------------
// one wave -> one 16x16 tile of XW; block has cols/16 waves sharing the X tile
// A and B fragments are pure aligned b128 bf16 loads (no in-loop conversion)
__global__ void xw_wmma_k(const __bf16* __restrict__ Xbf, const __bf16* __restrict__ QbfT,
                          float* __restrict__ XW, int cols) {
    int lane = threadIdx.x & 31;
    int wave = threadIdx.x >> 5;
    int m0 = blockIdx.x * 16;
    int n0 = wave * 16;
    int hi  = lane >> 4;      // 0 or 1
    int lml = lane & 15;
    int m = m0 + lml;         // A-fragment row for this lane
    int n = n0 + lml;         // B-fragment column for this lane
    v8f c = {};
#pragma unroll
    for (int k0 = 0; k0 < 128; k0 += 32) {
        // A: 16x32 bf16; lane<16: K = {k0..k0+7, k0+16..k0+23}; lane>=16: shifted by 8
        int kbase = k0 + (hi ? 8 : 0);
        const __bf16* xr = Xbf + (size_t)m * 128 + kbase;
        v8bf alo = *(const v8bf*)(xr);
        v8bf ahi = *(const v8bf*)(xr + 16);
        v16bf a = __builtin_shufflevector(alo, ahi, 0, 1, 2, 3, 4, 5, 6, 7,
                                          8, 9, 10, 11, 12, 13, 14, 15);
        // B: 32x16 bf16; same K order as A: lane<16: K = {k0..k0+7, k0+16..k0+23}; lane>=16: shifted by 8
        int koff = k0 + (hi ? 8 : 0);
        const __bf16* qr = QbfT + (size_t)n * 128 + koff;
        v8bf blo = *(const v8bf*)(qr);
        v8bf bhi = *(const v8bf*)(qr + 16);
        v16bf b = __builtin_shufflevector(blo, bhi, 0, 1, 2, 3, 4, 5, 6, 7,
                                          8, 9, 10, 11, 12, 13, 14, 15);
        c = __builtin_amdgcn_wmma_f32_16x16x32_bf16(false, a, false, b, (short)0, c, false, false);
    }
    // D layout: VGPR v -> M = v + (hi?8:0), N = lane&15
    int mb = m0 + (hi ? 8 : 0);
#pragma unroll
    for (int v = 0; v < 8; ++v)
        XW[(size_t)(mb + v) * cols + n] = c[v];
}

// ---------------- zero / scatter / relu ----------------
__global__ void zero4_k(float4* __restrict__ p, long n4) {
    long i = (long)blockIdx.x * blockDim.x + threadIdx.x;
    if (i < n4) p[i] = make_float4(0.f, 0.f, 0.f, 0.f);
}

__global__ void scatter_k(const float* __restrict__ XW, const int* __restrict__ src,
                          const int* __restrict__ dst, const float* __restrict__ w,
                          float* __restrict__ acc, int E_, int l2c, int cols) {
    long i = (long)blockIdx.x * blockDim.x + threadIdx.x;
    long total = (long)E_ << l2c;
    if (i >= total) return;
    int e = (int)(i >> l2c);
    int c = ((int)i & ((1 << l2c) - 1)) * 4;
    int s = src[e], d = dst[e];
    float ww = w[e];
    float4 xv = *(const float4*)(XW + (size_t)s * cols + c);
    float* ap = acc + (size_t)d * cols + c;
    atomicAdd(ap + 0, xv.x * ww);
    atomicAdd(ap + 1, xv.y * ww);
    atomicAdd(ap + 2, xv.z * ww);
    atomicAdd(ap + 3, xv.w * ww);
}

__global__ void relu_k(const float* __restrict__ a, float* __restrict__ o, long n) {
    long i = (long)blockIdx.x * blockDim.x + threadIdx.x;
    if (i < n) o[i] = fmaxf(a[i], 0.f);
}

// ---------------- host orchestration ----------------
extern "C" void kernel_launch(void* const* d_in, const int* in_sizes, int n_in,
                              void* d_out, int out_size, void* d_ws, size_t ws_size,
                              hipStream_t stream) {
    const int T = 4, N = 50000, E = 400000, F0 = 128, F1 = 128, F2 = 64;
    (void)in_sizes; (void)n_in; (void)out_size; (void)ws_size;

    const float* node_embs = (const float*)d_in[0];
    const float* mask      = (const float*)d_in[1];
    const int*   esrc      = (const int*)d_in[2];
    const int*   edst      = (const int*)d_in[3];
    const float* ew        = (const float*)d_in[4];
    const float* gcn_w0    = (const float*)d_in[5];
    const float* gcn_w1    = (const float*)d_in[6];
    const float* p0[10];
    const float* p1[10];
    for (int i = 0; i < 10; ++i) { p0[i] = (const float*)d_in[7 + i]; p1[i] = (const float*)d_in[17 + i]; }
    float* out = (float*)d_out;

    // workspace layout (256B aligned slices)
    char* wsb = (char*)d_ws;
    size_t off = 0;
    auto alloc = [&](size_t bytes) -> char* {
        char* p = wsb + off;
        off = (off + bytes + 255) & ~(size_t)255;
        return p;
    };
    float*  Q0a    = (float*)alloc((size_t)F0 * F1 * 4);
    float*  Q0b    = (float*)alloc((size_t)F0 * F1 * 4);
    float*  Q1a    = (float*)alloc((size_t)F1 * F2 * 4);
    float*  Q1b    = (float*)alloc((size_t)F1 * F2 * 4);
    __bf16* QbfT   = (__bf16*)alloc((size_t)128 * 128 * 2);
    float*  scores = (float*)alloc((size_t)N * 4);
    float*  vals   = (float*)alloc(128 * 4);
    int*    idxb   = (int*)alloc(128 * 4);
    float*  zbuf   = (float*)alloc((size_t)128 * 128 * 4);
    float*  invn   = (float*)alloc(256);
    float*  XWb    = (float*)alloc((size_t)N * 128 * 4);
    float*  accb   = (float*)alloc((size_t)N * 128 * 4);
    float*  h1     = (float*)alloc((size_t)N * 128 * 4);
    __bf16* Xbf    = (__bf16*)alloc((size_t)N * 128 * 2);

    // seed evolving GCN weights from inputs (fresh every launch -> deterministic)
    hipMemcpyAsync(Q0a, gcn_w0, (size_t)F0 * F1 * 4, hipMemcpyDeviceToDevice, stream);
    hipMemcpyAsync(Q1a, gcn_w1, (size_t)F1 * F2 * 4, hipMemcpyDeviceToDevice, stream);

    float* q0c = Q0a; float* q0n = Q0b;
    float* q1c = Q1a; float* q1n = Q1b;

    auto run_layer = [&](const float* X, const float* const* P, float*& qcur, float*& qnext,
                         int cols, float* outbuf, const float* mt, const int* st,
                         const int* dt_, const float* wt) {
        invnorm_k<<<1, 128, 0, stream>>>(P[9], invn);
        score_k<<<(N + 127) / 128, 128, 0, stream>>>(X, P[9], invn, mt, scores, N);
        topk_k<<<1, 1024, 0, stream>>>(scores, vals, idxb, N, cols);
        zbuild_k<<<(128 * cols + 255) / 256, 256, 0, stream>>>(X, idxb, vals, zbuf, cols);
        gru_k<<<cols, 128, 0, stream>>>(zbuf, qcur, P[0], P[1], P[2], P[3], P[4], P[5],
                                        P[6], P[7], P[8], qnext, QbfT, cols);
        long n8 = (long)N * 128 / 8;
        cvtbf_k<<<(unsigned)((n8 + 255) / 256), 256, 0, stream>>>(X, Xbf, n8);
        xw_wmma_k<<<N / 16, 32 * (cols / 16), 0, stream>>>(Xbf, QbfT, XWb, cols);
        long n4 = (long)N * cols / 4;
        zero4_k<<<(unsigned)((n4 + 255) / 256), 256, 0, stream>>>((float4*)accb, n4);
        int l2c = (cols == 128) ? 5 : 4;
        long tot = (long)E << l2c;
        scatter_k<<<(unsigned)((tot + 255) / 256), 256, 0, stream>>>(XWb, st, dt_, wt, accb, E, l2c, cols);
        long nel = (long)N * cols;
        relu_k<<<(unsigned)((nel + 255) / 256), 256, 0, stream>>>(accb, outbuf, nel);
        float* tmp = qcur; qcur = qnext; qnext = tmp;
    };

    for (int t = 0; t < T; ++t) {
        const float* Xt  = node_embs + (size_t)t * N * F0;
        const float* mt  = mask + (size_t)t * N;
        const int*   st  = esrc + (size_t)t * E;
        const int*   dt_ = edst + (size_t)t * E;
        const float* wt  = ew + (size_t)t * E;
        // layer 0: X_t -> h1 (cols = 128)
        run_layer(Xt, p0, q0c, q0n, F1, h1, mt, st, dt_, wt);
        // layer 1: h1 -> out[t] (cols = 64)
        run_layer(h1, p1, q1c, q1n, F2, out + (size_t)t * N * F2, mt, st, dt_, wt);
    }
}
